// StructureDistillLoss_50989851738278
// MI455X (gfx1250) — matrix operand order repaired and checked by reading, not compiled, not run on hardware
//
#include <hip/hip_runtime.h>
#include <hip/hip_bf16.h>

typedef __attribute__((ext_vector_type(16))) _Float16 v16h;
typedef __attribute__((ext_vector_type(8)))  float    v8f;

#define WIN     15
#define M_WIN   225          // WIN*WIN
#define M_PAD   240          // 15 tiles of 16
#define NTILE   15
#define NTRI    120          // NTILE*(NTILE+1)/2 upper-triangle tile pairs
#define CCH     64
#define FSTRIDE 66           // halves per row (33 dwords, gcd(33,64)=1 -> conflict-free)
#define HH      384
#define WW      512
#define BS      4
#define NANCH   100
#define EPSN    1e-8f

#define WMMA_F16(a, b, c) \
    __builtin_amdgcn_wmma_f32_16x16x32_f16(false, (a), false, (b), (short)0, (c), false, false)

// ---------------------------------------------------------------------------
__global__ void sdl_init(float* __restrict__ acc) {
    if (threadIdx.x < 8) acc[threadIdx.x] = 0.0f;
}

// ---------------------------------------------------------------------------
// one workgroup per (anchor n, batch b) window; 8 wave32s
// ---------------------------------------------------------------------------
__global__ __launch_bounds__(256) void sdl_main(
    const float* __restrict__ student,   // [bs,64,h,w]
    const float* __restrict__ teacher,   // [bs,64,h,w]
    const float* __restrict__ depth,     // [bs,1,h,w]
    const float* __restrict__ convw,     // [64,64]
    const float* __restrict__ convb,     // [64]
    const int*   __restrict__ coords,    // [bs,N,2]
    const int*   __restrict__ mind_p,
    const int*   __restrict__ maxd_p,
    float*       __restrict__ acc)       // [num0..3, den0..3]
{
    // Ft: first holds conv_w (rows 0..63) for the WMMA conv, then the
    // normalized teacher features. Fs: raw student, then conv output.
    __shared__ _Float16 Ft[M_PAD * FSTRIDE];
    __shared__ _Float16 Fs[M_PAD * FSTRIDE];
    __shared__ float    mg[M_PAD];
    __shared__ float    convb_s[CCH];
    __shared__ float    red[8];

    const int n    = blockIdx.x;
    const int b    = blockIdx.y;
    const int tid  = (int)threadIdx.x;
    const int lane = tid & 31;
    const int wv   = tid >> 5;
    const int l15  = lane & 15;
    const int lhi  = lane >> 4;
    const int hw   = HH * WW;

    const float mind = (float)mind_p[0];
    const float maxd = (float)maxd_p[0];

    // ---------------- Stage 1: staging ------------------------------------
    // zero pad rows 225..239 of both feature buffers + pad mask entries
    for (int i = tid; i < (M_PAD - M_WIN) * FSTRIDE; i += 256) {
        Ft[M_WIN * FSTRIDE + i] = (_Float16)0.0f;
        Fs[M_WIN * FSTRIDE + i] = (_Float16)0.0f;
    }
    if (tid < M_PAD - M_WIN) mg[M_WIN + tid] = 0.0f;
    // conv weights -> Ft rows 0..63 (f16), bias -> LDS
    for (int i = tid; i < CCH * CCH; i += 256) {
        const int r = i >> 6, c = i & 63;
        Ft[r * FSTRIDE + c] = (_Float16)convw[i];
    }
    if (tid < CCH) convb_s[tid] = convb[tid];

    int pix = 0;
    if (tid < M_WIN) {
        const int m  = tid;
        const int dy = m / WIN;
        const int dx = m - dy * WIN;
        const int x0 = coords[((b * NANCH + n) << 1) + 0];
        const int y0 = coords[((b * NANCH + n) << 1) + 1];
        pix = (y0 + dy) * WW + (x0 + dx);

        const float d = depth[(size_t)b * hw + pix];
        mg[m] = (d > mind && d < maxd) ? 1.0f : 0.0f;

        // raw (unmasked) student pixels -> Fs[m][c]  (mask applies post-conv)
        const float* sp = student + (size_t)b * CCH * hw + pix;
        #pragma unroll
        for (int c = 0; c < CCH; ++c)
            Fs[m * FSTRIDE + c] = (_Float16)sp[c * hw];
    }
    __syncthreads();

    // ---------------- Stage 2: 1x1 conv as WMMA GEMM ----------------------
    // D[ch, pixel] = W[ch, c] * Sraw[c, pixel] ; 4 ch-tiles x 15 pixel-tiles
    v8f Cacc[8];
    #pragma unroll
    for (int it = 0; it < 8; ++it) {
        const int t = wv + (it << 3);          // wave-uniform guard
        if (t < 4 * NTILE) {
            const int R  = (t / NTILE) << 4;   // output-channel block
            const int Cc = (t % NTILE) << 4;   // pixel block
            const _Float16* ap = Ft + (R + l15) * FSTRIDE + (lhi << 3);
            const _Float16* bp = Fs + (Cc + l15) * FSTRIDE + (lhi << 3);
            v16h a0, a1, b0, b1;
            #pragma unroll
            for (int e = 0; e < 8; ++e) {
                a0[e] = ap[e];      a0[8 + e] = ap[16 + e];
                a1[e] = ap[32 + e]; a1[8 + e] = ap[48 + e];
            }
            #pragma unroll
            for (int e = 0; e < 16; ++e) { b0[e] = bp[e + ((e >> 3) << 3)]; b1[e] = bp[32 + e + ((e >> 3) << 3)]; }
            v8f c = {};
            c = WMMA_F16(a0, b0, c);
            c = WMMA_F16(a1, b1, c);
            Cacc[it] = c;
        }
    }
    __syncthreads();   // all conv reads of Fs(raw) / Ft(W) complete

    // ---------------- Stage 3: writeback conv (bias+mask) & teacher -------
    #pragma unroll
    for (int it = 0; it < 8; ++it) {
        const int t = wv + (it << 3);
        if (t < 4 * NTILE) {
            const int R   = (t / NTILE) << 4;
            const int Cc  = (t % NTILE) << 4;
            const int p   = Cc + l15;              // pixel (column of D)
            const float mv = mg[p];
            _Float16* dst = Fs + p * FSTRIDE + R + (lhi << 3);
            const v8f c = Cacc[it];
            #pragma unroll
            for (int v = 0; v < 8; ++v) {
                const int ch = R + (lhi << 3) + v; // row of D
                dst[v] = (_Float16)((c[v] + convb_s[ch]) * mv);
            }
        }
    }
    // teacher: gather, mask, normalize -> Ft (overwrites W region)
    if (tid < M_WIN) {
        const int m = tid;
        const float mv = mg[m];
        const float* tp = teacher + (size_t)b * CCH * hw + pix;
        float ss = 0.0f;
        #pragma unroll
        for (int c = 0; c < CCH; ++c) {
            const float tv = tp[c * hw] * mv;
            ss += tv * tv;
            Ft[m * FSTRIDE + c] = (_Float16)tv;
        }
        const float inv = 1.0f / (sqrtf(ss) + EPSN);
        #pragma unroll
        for (int c = 0; c < CCH; ++c)
            Ft[m * FSTRIDE + c] = (_Float16)((float)Ft[m * FSTRIDE + c] * inv);
    }
    __syncthreads();

    // ---------------- Stage 4: normalize conv output columns --------------
    if (tid < M_WIN) {
        const int m = tid;
        float ss = 0.0f;
        #pragma unroll
        for (int c = 0; c < CCH; ++c) {
            const float v = (float)Fs[m * FSTRIDE + c];
            ss += v * v;
        }
        const float inv = 1.0f / (sqrtf(ss) + EPSN);
        #pragma unroll
        for (int c = 0; c < CCH; ++c)
            Fs[m * FSTRIDE + c] = (_Float16)((float)Fs[m * FSTRIDE + c] * inv);
    }
    __syncthreads();

    // ---------------- Stage 5: WMMA Gram tiles (upper triangle) -----------
    // T and S are symmetric -> (T-S)^2 and the mask product are symmetric.
    // Sum over full matrix = diag tiles x1 + off-diag upper tiles x2.
    // 120 tile pairs / 8 waves = 15 per wave, perfectly balanced.
    float lsum = 0.0f;
    for (int t = wv; t < NTRI; t += 8) {            // wave-uniform trip count
        // decode upper-triangle index t -> (i, j), i <= j (uniform scalar loop)
        int idx = t, i = 0;
        while (idx >= NTILE - i) { idx -= NTILE - i; ++i; }
        const int j  = i + idx;
        const int R  = i << 4;                      // row block
        const int Cc = j << 4;                      // col block

        const _Float16* apT = Ft + (R + l15) * FSTRIDE + (lhi << 3);
        const _Float16* apS = Fs + (R + l15) * FSTRIDE + (lhi << 3);
        const _Float16* bpT = Ft + (Cc + l15) * FSTRIDE + (lhi << 3);
        const _Float16* bpS = Fs + (Cc + l15) * FSTRIDE + (lhi << 3);

        v16h aT0, aT1, aS0, aS1, bT0, bT1, bS0, bS1;
        #pragma unroll
        for (int e = 0; e < 8; ++e) {
            aT0[e] = apT[e];        aT0[8 + e] = apT[16 + e];
            aT1[e] = apT[32 + e];   aT1[8 + e] = apT[48 + e];
            aS0[e] = apS[e];        aS0[8 + e] = apS[16 + e];
            aS1[e] = apS[32 + e];   aS1[8 + e] = apS[48 + e];
        }
        #pragma unroll
        for (int e = 0; e < 16; ++e) {
            bT0[e] = bpT[e + ((e >> 3) << 3)];        bT1[e] = bpT[32 + e + ((e >> 3) << 3)];
            bS0[e] = bpS[e + ((e >> 3) << 3)];        bS1[e] = bpS[32 + e + ((e >> 3) << 3)];
        }

        v8f cT = {};
        v8f cS = {};
        cT = WMMA_F16(aT0, bT0, cT);
        cT = WMMA_F16(aT1, bT1, cT);
        cS = WMMA_F16(aS0, bS0, cS);
        cS = WMMA_F16(aS1, bS1, cS);

        // C/D: lane -> col Cc+(lane&15); vgpr v -> row R + v + 8*(lane>=16)
        const float mk    = mg[Cc + l15];
        const int   mbase = R + (lhi << 3);
        float tsum = 0.0f;
        #pragma unroll
        for (int v = 0; v < 8; ++v) {
            const float dd = cT[v] - cS[v];
            tsum += mg[mbase + v] * dd * dd;
        }
        const float wgt = (i == j) ? 1.0f : 2.0f;   // off-diagonal counted twice
        lsum += wgt * mk * tsum;
    }

    // ---------------- reduce: wave32 -> WG -> global atomics --------------
    #pragma unroll
    for (int o = 16; o > 0; o >>= 1) lsum += __shfl_xor(lsum, o, 32);
    if (lane == 0) red[wv] = lsum;
    __syncthreads();

    if (tid == 0) {
        float tot = 0.0f;
        #pragma unroll
        for (int i = 0; i < 8; ++i) tot += red[i];
        float msum = 0.0f;
        for (int m = 0; m < M_WIN; ++m) msum += mg[m];
        atomicAdd(&acc[b], tot);             // numerator for batch b
        atomicAdd(&acc[4 + b], msum * msum); // denominator for batch b
    }
}

// ---------------------------------------------------------------------------
__global__ void sdl_final(const float* __restrict__ acc, float* __restrict__ out) {
    if (threadIdx.x == 0) {
        float s = 0.0f;
        #pragma unroll
        for (int b = 0; b < BS; ++b) s += acc[b] / acc[4 + b];
        out[0] = s * (1.0f / (float)BS);
    }
}

// ---------------------------------------------------------------------------
extern "C" void kernel_launch(void* const* d_in, const int* in_sizes, int n_in,
                              void* d_out, int out_size, void* d_ws, size_t ws_size,
                              hipStream_t stream) {
    const float* student = (const float*)d_in[0];
    const float* teacher = (const float*)d_in[1];
    const float* depth   = (const float*)d_in[2];
    const float* convw   = (const float*)d_in[3];
    const float* convb   = (const float*)d_in[4];
    const int*   coords  = (const int*)d_in[5];
    const int*   mind_p  = (const int*)d_in[6];
    const int*   maxd_p  = (const int*)d_in[7];
    float*       acc     = (float*)d_ws;     // 8 floats: num[4], den[4]
    float*       out     = (float*)d_out;

    sdl_init<<<1, 32, 0, stream>>>(acc);
    sdl_main<<<dim3(NANCH, BS), 256, 0, stream>>>(
        student, teacher, depth, convw, convb, coords, mind_p, maxd_p, acc);
    sdl_final<<<1, 32, 0, stream>>>(acc, out);
}
